// UnidirectionalAttention_90890097918077
// MI455X (gfx1250) — compile-verified, matrix operand order checked by reading, not run on hardware
//
#include <hip/hip_runtime.h>
#include <hip/hip_bf16.h>
#include <math.h>

// Problem constants (from reference): B=2, S=2048, D=2048, H=16, DH=128
#define BATCH 2
#define SEQ   2048
#define DMODEL 2048
#define NHEAD 16
#define HDIM  128
#define NQKV  (3 * NHEAD * HDIM)          // 6144
#define MROWS (BATCH * SEQ)               // 4096
#define ATTN_SCALE 0.08838834764831845f   // 1/sqrt(128)

typedef __attribute__((ext_vector_type(16))) _Float16 v16h;
typedef __attribute__((ext_vector_type(8)))  _Float16 v8h;
typedef __attribute__((ext_vector_type(4)))  _Float16 v4h;
typedef __attribute__((ext_vector_type(8)))  float    v8f;

// ---------------------------------------------------------------------------
// WMMA helper: D = A(16x32 f16) * B(32x16 f16) + C(16x16 f32)
// ---------------------------------------------------------------------------
__device__ __forceinline__ v8f wmma32(v16h a, v16h b, v8f c) {
  return __builtin_amdgcn_wmma_f32_16x16x32_f16(
      /*neg_a=*/false, a, /*neg_b=*/false, b,
      /*c_mod=*/(short)0, c, /*reuse_a=*/false, /*reuse_b=*/false);
}

// ---------------------------------------------------------------------------
// Fragment loader (ISA 16-bit A/B layout, 16x32, K contiguous in memory):
//   lane l owns row/col  = l & 15
//   halves 0..7  -> K = kb+0..kb+7,  halves 8..15 -> K = 16+kb+0..7
//   where kb = (l < 16) ? 0 : 8.   Two 16-byte loads per fragment.
// ---------------------------------------------------------------------------
__device__ __forceinline__ v16h load_frag(const _Float16* tile, int ld) {
  const int lane = threadIdx.x & 31;
  const int r  = lane & 15;
  const int kb = (lane & 16) >> 1;     // 0 or 8
  const _Float16* p = tile + r * ld + kb;
  v8h lo = *(const v8h*)(p);
  v8h hi = *(const v8h*)(p + 16);
  v16h f;
#pragma unroll
  for (int i = 0; i < 8; ++i) { f[i] = lo[i]; f[i + 8] = hi[i]; }
  return f;
}

// ---------------------------------------------------------------------------
// fp32 -> f16 conversion, 4 elements / thread
// ---------------------------------------------------------------------------
__global__ __launch_bounds__(256) void cvt_f32_f16(const float4* __restrict__ in,
                                                   v4h* __restrict__ out,
                                                   int n4) {
  int i = blockIdx.x * 256 + threadIdx.x;
  if (i < n4) {
    float4 v = in[i];
    v4h o;
    o[0] = (_Float16)v.x; o[1] = (_Float16)v.y;
    o[2] = (_Float16)v.z; o[3] = (_Float16)v.w;
    out[i] = o;
  }
}

// ---------------------------------------------------------------------------
// QKV projection:  C[m, o] = sum_k X[m,k] * Wqkv[o,k] + b[o]
// Wave tile: 32(M) x 64(N), software-pipelined K loop (step 32).
// Epilogue scatters into Q,K head-major (B,H,S,DH) and V^T (B,H,DH,S), f16.
// ---------------------------------------------------------------------------
__global__ __launch_bounds__(256) void qkv_gemm(const _Float16* __restrict__ X,
                                                const _Float16* __restrict__ W,
                                                const float* __restrict__ bias,
                                                _Float16* __restrict__ Q,
                                                _Float16* __restrict__ K,
                                                _Float16* __restrict__ VT) {
  const int wave = (blockIdx.x * blockDim.x + threadIdx.x) >> 5;
  const int NT = NQKV / 64;                  // 96 n-strips
  const int mt = wave / NT;
  const int nt = wave % NT;
  const int m0 = mt * 32;
  const int n0 = nt * 64;
  const int lane = threadIdx.x & 31;

  const _Float16* Xrow = X + (size_t)m0 * DMODEL;
  const _Float16* Wrow = W + (size_t)n0 * DMODEL;

  v8f acc[8];
#pragma unroll
  for (int i = 0; i < 8; ++i) acc[i] = (v8f){};

  // preload k0 = 0
  v16h a0 = load_frag(Xrow, DMODEL);
  v16h a1 = load_frag(Xrow + 16 * DMODEL, DMODEL);
  v16h b0 = load_frag(Wrow + 0 * 16 * DMODEL, DMODEL);
  v16h b1 = load_frag(Wrow + 1 * 16 * DMODEL, DMODEL);
  v16h b2 = load_frag(Wrow + 2 * 16 * DMODEL, DMODEL);
  v16h b3 = load_frag(Wrow + 3 * 16 * DMODEL, DMODEL);

  for (int k0 = 32; k0 < DMODEL; k0 += 32) {
    // prefetch next K step while WMMAs consume current registers
    v16h a0n = load_frag(Xrow + k0, DMODEL);
    v16h a1n = load_frag(Xrow + 16 * DMODEL + k0, DMODEL);
    v16h b0n = load_frag(Wrow + 0 * 16 * DMODEL + k0, DMODEL);
    v16h b1n = load_frag(Wrow + 1 * 16 * DMODEL + k0, DMODEL);
    v16h b2n = load_frag(Wrow + 2 * 16 * DMODEL + k0, DMODEL);
    v16h b3n = load_frag(Wrow + 3 * 16 * DMODEL + k0, DMODEL);

    acc[0] = wmma32(a0, b0, acc[0]);
    acc[1] = wmma32(a0, b1, acc[1]);
    acc[2] = wmma32(a0, b2, acc[2]);
    acc[3] = wmma32(a0, b3, acc[3]);
    acc[4] = wmma32(a1, b0, acc[4]);
    acc[5] = wmma32(a1, b1, acc[5]);
    acc[6] = wmma32(a1, b2, acc[6]);
    acc[7] = wmma32(a1, b3, acc[7]);

    a0 = a0n; a1 = a1n; b0 = b0n; b1 = b1n; b2 = b2n; b3 = b3n;
  }
  // drain
  acc[0] = wmma32(a0, b0, acc[0]);
  acc[1] = wmma32(a0, b1, acc[1]);
  acc[2] = wmma32(a0, b2, acc[2]);
  acc[3] = wmma32(a0, b3, acc[3]);
  acc[4] = wmma32(a1, b0, acc[4]);
  acc[5] = wmma32(a1, b1, acc[5]);
  acc[6] = wmma32(a1, b2, acc[6]);
  acc[7] = wmma32(a1, b3, acc[7]);

  const int c = lane & 15;
  const int rbase = (lane & 16) >> 1;        // 0 or 8
#pragma unroll
  for (int ms = 0; ms < 2; ++ms) {
#pragma unroll
    for (int i = 0; i < 4; ++i) {
      const int o = n0 + i * 16 + c;
      const float bv = bias[o];
      const int part = o >> 11;              // 0=q,1=k,2=v
      const int rem  = o & 2047;
      const int h  = rem >> 7;
      const int dh = rem & 127;
#pragma unroll
      for (int r = 0; r < 8; ++r) {
        const int row = m0 + ms * 16 + rbase + r;   // row = b*S + s
        const int bb = row >> 11;
        const int s  = row & 2047;
        const int bh = bb * NHEAD + h;
        const _Float16 val = (_Float16)(acc[ms * 4 + i][r] + bv);
        if (part == 0)      Q[((size_t)bh * SEQ + s) * HDIM + dh] = val;
        else if (part == 1) K[((size_t)bh * SEQ + s) * HDIM + dh] = val;
        else                VT[((size_t)bh * HDIM + dh) * SEQ + s] = val;
      }
    }
  }
}

// ---------------------------------------------------------------------------
// Flash attention (causal). One wave per (b,h, 16-row q tile).
// K fragments batched before score WMMAs; V fragments hoisted so the softmax
// VALU chain hides their latency. P staged via LDS (C layout -> A layout).
// Output: attn (B, S, H*DH) f16.
// ---------------------------------------------------------------------------
__global__ __launch_bounds__(256) void flash_attn(const _Float16* __restrict__ Q,
                                                  const _Float16* __restrict__ K,
                                                  const _Float16* __restrict__ VT,
                                                  _Float16* __restrict__ O) {
  __shared__ __align__(16) _Float16 pshare[8][16 * 32];

  const int wid  = (blockIdx.x * blockDim.x + threadIdx.x) >> 5;
  const int lane = threadIdx.x & 31;
  const int wslot = threadIdx.x >> 5;
  const int bh = wid >> 7;                   // / (SEQ/16)
  const int qt = wid & 127;
  const int q0 = qt << 4;

  // Resident Q fragments (16 x 128)
  const _Float16* qbase = Q + ((size_t)bh * SEQ + q0) * HDIM;
  v16h qf[4];
#pragma unroll
  for (int i = 0; i < 4; ++i) qf[i] = load_frag(qbase + i * 32, HDIM);

  v8f oacc[8];
#pragma unroll
  for (int d = 0; d < 8; ++d) oacc[d] = (v8f){};

  float m[8], lsum[8];
#pragma unroll
  for (int r = 0; r < 8; ++r) { m[r] = -INFINITY; lsum[r] = 0.0f; }

  const int c = lane & 15;
  const int rbase = (lane & 16) >> 1;
  const int jt_max = (q0 + 15) >> 5;

  for (int jt = 0; jt <= jt_max; ++jt) {
    const int j0 = jt << 5;
    const _Float16* kb0 = K + ((size_t)bh * SEQ + j0) * HDIM;

    // --- batch-load all 8 K fragments, then 8 back-to-back score WMMAs ---
    v16h kf[8];
#pragma unroll
    for (int kk = 0; kk < 4; ++kk) {
      kf[kk]     = load_frag(kb0 + kk * 32, HDIM);
      kf[4 + kk] = load_frag(kb0 + 16 * HDIM + kk * 32, HDIM);
    }
    v8f s0 = (v8f){}, s1 = (v8f){};
#pragma unroll
    for (int kk = 0; kk < 4; ++kk) s0 = wmma32(qf[kk], kf[kk], s0);
#pragma unroll
    for (int kk = 0; kk < 4; ++kk) s1 = wmma32(qf[kk], kf[4 + kk], s1);

    // --- hoist V fragment loads; softmax VALU below hides their latency ---
    v16h vf[8];
    const _Float16* vb = VT + (size_t)bh * HDIM * SEQ + j0;
#pragma unroll
    for (int d = 0; d < 8; ++d) vf[d] = load_frag(vb + (size_t)(d * 16) * SEQ, SEQ);

    // --- causal mask + online softmax ---
    float p0[8], p1[8], alpha[8];
#pragma unroll
    for (int r = 0; r < 8; ++r) {
      const int rowg = q0 + rbase + r;
      float v0 = (j0 + c <= rowg)      ? s0[r] * ATTN_SCALE : -INFINITY;
      float v1 = (j0 + 16 + c <= rowg) ? s1[r] * ATTN_SCALE : -INFINITY;
      float mx = fmaxf(v0, v1);
      mx = fmaxf(mx, __shfl_xor(mx, 1, 16));
      mx = fmaxf(mx, __shfl_xor(mx, 2, 16));
      mx = fmaxf(mx, __shfl_xor(mx, 4, 16));
      mx = fmaxf(mx, __shfl_xor(mx, 8, 16));
      const float nm = fmaxf(m[r], mx);
      alpha[r] = __expf(m[r] - nm);
      m[r] = nm;
      p0[r] = __expf(v0 - nm);
      p1[r] = __expf(v1 - nm);
      float rs = p0[r] + p1[r];
      rs += __shfl_xor(rs, 1, 16);
      rs += __shfl_xor(rs, 2, 16);
      rs += __shfl_xor(rs, 4, 16);
      rs += __shfl_xor(rs, 8, 16);
      lsum[r] = lsum[r] * alpha[r] + rs;
    }

    // rescale running output
#pragma unroll
    for (int d = 0; d < 8; ++d)
#pragma unroll
      for (int r = 0; r < 8; ++r) oacc[d][r] *= alpha[r];

    // --- stage P (16x32 f16) through LDS: C layout -> A layout ---
    _Float16* pp = pshare[wslot];
#pragma unroll
    for (int r = 0; r < 8; ++r) {
      pp[(rbase + r) * 32 + c]      = (_Float16)p0[r];
      pp[(rbase + r) * 32 + 16 + c] = (_Float16)p1[r];
    }
    asm volatile("s_wait_dscnt 0" ::: "memory");
    v16h pf = load_frag(pp, 32);

    // --- O += P * V ---
#pragma unroll
    for (int d = 0; d < 8; ++d) oacc[d] = wmma32(pf, vf[d], oacc[d]);
  }

  // --- writeback: attn (B, S, H*DH) f16 ---
  const int bb = bh >> 4;
  const int h  = bh & 15;
#pragma unroll
  for (int d = 0; d < 8; ++d)
#pragma unroll
    for (int r = 0; r < 8; ++r) {
      const float v = oacc[d][r] / lsum[r];
      O[((size_t)bb * SEQ + q0 + rbase + r) * DMODEL + h * HDIM + d * 16 + c] =
          (_Float16)v;
    }
}

// ---------------------------------------------------------------------------
// Output projection: out[m, n] = sum_k Attn[m,k] * Wout[n,k] + b[n] (fp32 out)
// Wave tile: 32(M) x 64(N), software-pipelined K loop.
// ---------------------------------------------------------------------------
__global__ __launch_bounds__(256) void out_gemm(const _Float16* __restrict__ A,
                                                const _Float16* __restrict__ W,
                                                const float* __restrict__ bias,
                                                float* __restrict__ out) {
  const int wave = (blockIdx.x * blockDim.x + threadIdx.x) >> 5;
  const int NT = DMODEL / 64;                // 32 n-strips
  const int mt = wave / NT;
  const int nt = wave % NT;
  const int m0 = mt * 32;
  const int n0 = nt * 64;
  const int lane = threadIdx.x & 31;

  const _Float16* Arow = A + (size_t)m0 * DMODEL;
  const _Float16* Wrow = W + (size_t)n0 * DMODEL;

  v8f acc[8];
#pragma unroll
  for (int i = 0; i < 8; ++i) acc[i] = (v8f){};

  v16h a0 = load_frag(Arow, DMODEL);
  v16h a1 = load_frag(Arow + 16 * DMODEL, DMODEL);
  v16h b0 = load_frag(Wrow + 0 * 16 * DMODEL, DMODEL);
  v16h b1 = load_frag(Wrow + 1 * 16 * DMODEL, DMODEL);
  v16h b2 = load_frag(Wrow + 2 * 16 * DMODEL, DMODEL);
  v16h b3 = load_frag(Wrow + 3 * 16 * DMODEL, DMODEL);

  for (int k0 = 32; k0 < DMODEL; k0 += 32) {
    v16h a0n = load_frag(Arow + k0, DMODEL);
    v16h a1n = load_frag(Arow + 16 * DMODEL + k0, DMODEL);
    v16h b0n = load_frag(Wrow + 0 * 16 * DMODEL + k0, DMODEL);
    v16h b1n = load_frag(Wrow + 1 * 16 * DMODEL + k0, DMODEL);
    v16h b2n = load_frag(Wrow + 2 * 16 * DMODEL + k0, DMODEL);
    v16h b3n = load_frag(Wrow + 3 * 16 * DMODEL + k0, DMODEL);

    acc[0] = wmma32(a0, b0, acc[0]);
    acc[1] = wmma32(a0, b1, acc[1]);
    acc[2] = wmma32(a0, b2, acc[2]);
    acc[3] = wmma32(a0, b3, acc[3]);
    acc[4] = wmma32(a1, b0, acc[4]);
    acc[5] = wmma32(a1, b1, acc[5]);
    acc[6] = wmma32(a1, b2, acc[6]);
    acc[7] = wmma32(a1, b3, acc[7]);

    a0 = a0n; a1 = a1n; b0 = b0n; b1 = b1n; b2 = b2n; b3 = b3n;
  }
  acc[0] = wmma32(a0, b0, acc[0]);
  acc[1] = wmma32(a0, b1, acc[1]);
  acc[2] = wmma32(a0, b2, acc[2]);
  acc[3] = wmma32(a0, b3, acc[3]);
  acc[4] = wmma32(a1, b0, acc[4]);
  acc[5] = wmma32(a1, b1, acc[5]);
  acc[6] = wmma32(a1, b2, acc[6]);
  acc[7] = wmma32(a1, b3, acc[7]);

  const int c = lane & 15;
  const int rbase = (lane & 16) >> 1;
#pragma unroll
  for (int ms = 0; ms < 2; ++ms) {
#pragma unroll
    for (int i = 0; i < 4; ++i) {
      const int n = n0 + i * 16 + c;
      const float bv = bias[n];
#pragma unroll
      for (int r = 0; r < 8; ++r) {
        const int row = m0 + ms * 16 + rbase + r;
        out[(size_t)row * DMODEL + n] = acc[ms * 4 + i][r] + bv;
      }
    }
  }
}

// ---------------------------------------------------------------------------
// Host launcher
// ---------------------------------------------------------------------------
extern "C" void kernel_launch(void* const* d_in, const int* in_sizes, int n_in,
                              void* d_out, int out_size, void* d_ws, size_t ws_size,
                              hipStream_t stream) {
  const float* x     = (const float*)d_in[0];   // (B,S,D)
  const float* w_qkv = (const float*)d_in[1];   // (3*H*DH, D)
  const float* b_qkv = (const float*)d_in[2];   // (3*H*DH)
  const float* w_out = (const float*)d_in[3];   // (D, H*DH)
  const float* b_out = (const float*)d_in[4];   // (D)
  float* out = (float*)d_out;

  const int nx    = MROWS * DMODEL;   // 8,388,608
  const int nwqkv = NQKV * DMODEL;    // 12,582,912
  const int nwout = DMODEL * DMODEL;  // 4,194,304
  const int nhead = BATCH * NHEAD * SEQ * HDIM;  // 8,388,608

  _Float16* x16    = (_Float16*)d_ws;
  _Float16* wqkv16 = x16 + nx;
  _Float16* wout16 = wqkv16 + nwqkv;
  _Float16* q16    = wout16 + nwout;
  _Float16* k16    = q16 + nhead;
  _Float16* vT16   = k16 + nhead;
  _Float16* attn16 = vT16 + nhead;    // MROWS*DMODEL halves

  cvt_f32_f16<<<(nx / 4 + 255) / 256, 256, 0, stream>>>(
      (const float4*)x, (v4h*)x16, nx / 4);
  cvt_f32_f16<<<(nwqkv / 4 + 255) / 256, 256, 0, stream>>>(
      (const float4*)w_qkv, (v4h*)wqkv16, nwqkv / 4);
  cvt_f32_f16<<<(nwout / 4 + 255) / 256, 256, 0, stream>>>(
      (const float4*)w_out, (v4h*)wout16, nwout / 4);

  // (MROWS/32) * (NQKV/64) waves = 128*96 = 12288 -> 1536 blocks of 8 waves
  qkv_gemm<<<1536, 256, 0, stream>>>(x16, wqkv16, b_qkv, q16, k16, vT16);

  // B*H*(S/16) = 4096 waves -> 512 blocks
  flash_attn<<<512, 256, 0, stream>>>(q16, k16, vT16, attn16);

  // (MROWS/32) * (DMODEL/64) waves = 128*32 = 4096 -> 512 blocks
  out_gemm<<<512, 256, 0, stream>>>(attn16, wout16, b_out, out);
}
